// AttentionReadout_59442347376786
// MI455X (gfx1250) — matrix operand order repaired and checked by reading, not compiled, not run on hardware
//
#include <hip/hip_runtime.h>
#include <hip/hip_bf16.h>
#include <math.h>

typedef __bf16 bf16;
typedef __attribute__((ext_vector_type(16))) __bf16 v16bf;
typedef __attribute__((ext_vector_type(8)))  __bf16 v8bf;
typedef __attribute__((ext_vector_type(8)))  float  v8f;

#define DD   256
#define GG   1024
#define NPB  512        // nodes per block in pooling kernel
#define WPAD 40         // padded row pitch (bf16 elems) of transposed W1 tiles

#if __has_builtin(__builtin_amdgcn_tanhf)
#define FAST_TANH(v) __builtin_amdgcn_tanhf(v)
#else
#define FAST_TANH(v) tanhf(v)
#endif

// sched_group_barrier masks
#define SGB_WMMA 0x008
#define SGB_DSRD 0x100

// ---------------------------------------------------------------------------
// One-time prep: W1 fp32 row-major [k][n] -> bf16 transposed padded tiles
// Wt[kt][n][kk], kt=0..7, n=0..255, kk=0..31, row pitch WPAD.
// Lane fragment reads become 32 contiguous bytes (2x ds_load_b128).
// ---------------------------------------------------------------------------
__global__ void k_prep(const float* __restrict__ W1, bf16* __restrict__ Wt)
{
    int i = blockIdx.x * 256 + threadIdx.x;   // i = k*256 + n  (coalesced read)
    if (i < 256 * 256) {
        int k  = i >> 8;
        int nn = i & 255;
        int kt = k >> 5;
        int kk = k & 31;
        Wt[kt * (256 * WPAD) + nn * WPAD + kk] = (bf16)W1[i];
    }
}

// B fragment: b[j] = W1[kbase + 16*half + j][col], j=0..15 -> 32 contiguous B.
__device__ __forceinline__ v16bf load_bfrag(const bf16* __restrict__ base,
                                            int col, int half)
{
    const v8bf* bp = reinterpret_cast<const v8bf*>(base + col * WPAD + half * 16);
    v8bf blo = bp[0];
    v8bf bhi = bp[1];
    return __builtin_shufflevector(
        blo, bhi, 0, 1, 2, 3, 4, 5, 6, 7, 8, 9, 10, 11, 12, 13, 14, 15);
}

// ---------------------------------------------------------------------------
// Kernel 1: scores[i] = tanh(x[i,:] @ W1 + b1) @ W2 + b2
// Block = 256 threads = 8 waves; each wave computes 16 rows -> 128 rows/block.
// Double-buffered LDS tiles (stage kt+1 during compute of kt) + software-
// pipelined B fragments, enforced with sched_group_barrier so the register
// allocator must keep two fragment pairs live (dscnt<=2 instead of 0).
// ---------------------------------------------------------------------------
__launch_bounds__(256)
__global__ void k_scores(const float* __restrict__ x,
                         const bf16*  __restrict__ Wt,
                         const float* __restrict__ b1,
                         const float* __restrict__ W2,
                         const float* __restrict__ b2,
                         float* __restrict__ scores,
                         int n)
{
    __shared__ __align__(16) bf16 sWt[2][256 * WPAD];   // 2 x 20 KB

    const int tid  = threadIdx.x;
    const int lane = tid & 31;
    const int wave = tid >> 5;
    const int half = lane >> 4;     // 0: lanes 0-15, 1: lanes 16-31
    const int l15  = lane & 15;

    const int rowBase = blockIdx.x * 128 + wave * 16;

    v8f acc[16] = {};               // 16 C-tiles: full 256-wide h row block

    // Row this lane supplies for the A fragment (clamped; writes are guarded).
    int arow = rowBase + l15;
    if (arow > n - 1) arow = n - 1;
    const float2* xrow = reinterpret_cast<const float2*>(x + (size_t)arow * DD);

    constexpr int CPT = 256 * WPAD * 2 / 16 / 256;   // 16B chunks per thread (5)

    // Prologue: stage tile 0 into buffer 0.
    {
        const uint4* src = reinterpret_cast<const uint4*>(Wt);
        uint4* dst = reinterpret_cast<uint4*>(sWt[0]);
#pragma unroll
        for (int j = 0; j < CPT; ++j)
            dst[tid + j * 256] = src[tid + j * 256];
    }
    __syncthreads();

    for (int kt = 0; kt < 8; ++kt) {
        const int cur = kt & 1;

        // Stage tile kt+1 into the alternate buffer while computing on cur.
        if (kt < 7) {
            const uint4* src = reinterpret_cast<const uint4*>(
                Wt + (size_t)(kt + 1) * (256 * WPAD));
            uint4* dst = reinterpret_cast<uint4*>(sWt[cur ^ 1]);
#pragma unroll
            for (int j = 0; j < CPT; ++j)
                dst[tid + j * 256] = src[tid + j * 256];
        }

        // A fragment: 16x32 bf16. Lane L holds row rowBase+L; lanes L and
        // L+16 split the 32-wide K slice (one full 128B line per row).
        v16bf a;
#pragma unroll
        for (int v = 0; v < 8; ++v) {
            int kk = 2 * v + half * 16;
            float2 p = xrow[(kt * 32 + kk) >> 1];
            a[2 * v]     = (bf16)p.x;
            a[2 * v + 1] = (bf16)p.y;
        }

        // 16 N-tiles, software-pipelined B fragments (2 in flight).
        const bf16* bbase = sWt[cur];
        v16bf bfrag[2];
        bfrag[0] = load_bfrag(bbase, l15, half);            // nt = 0
#pragma unroll
        for (int nt = 0; nt < 16; ++nt) {
            if (nt < 15)
                bfrag[(nt + 1) & 1] =
                    load_bfrag(bbase, (nt + 1) * 16 + l15, half);
            acc[nt] = __builtin_amdgcn_wmma_f32_16x16x32_bf16(
                false, a, false, bfrag[nt & 1], (short)0, acc[nt],
                false, false);
        }

        // Enforce the pipelined schedule: 4 leading ds-reads (fragments 0+1),
        // then [1 WMMA, 2 ds-reads] x 14, then the last 2 WMMAs. This makes
        // the two bfrag lifetimes overlap -> distinct registers -> dscnt<=2.
        __builtin_amdgcn_sched_group_barrier(SGB_DSRD, 4, 0);
#pragma unroll
        for (int g = 0; g < 14; ++g) {
            __builtin_amdgcn_sched_group_barrier(SGB_WMMA, 1, 0);
            __builtin_amdgcn_sched_group_barrier(SGB_DSRD, 2, 0);
        }
        __builtin_amdgcn_sched_group_barrier(SGB_WMMA, 2, 0);

        __syncthreads();   // staging of kt+1 done; reads of cur done
    }

    // Epilogue: h = tanh(acc + b1); partial score = h .* W2 summed over N.
    // C layout: lanes 0-15 hold (M=v, N=l15+16*nt); lanes 16-31 hold M=v+8.
    float partial[8];
#pragma unroll
    for (int v = 0; v < 8; ++v) partial[v] = 0.0f;

#pragma unroll
    for (int nt = 0; nt < 16; ++nt) {
        const int col = nt * 16 + l15;
        const float bb = b1[col];
        const float w2 = W2[col];
#pragma unroll
        for (int v = 0; v < 8; ++v) {
            float h = FAST_TANH(acc[nt][v] + bb);
            partial[v] += h * w2;
        }
    }
    // Reduce across the 16 lanes of each half (masks < 16 stay in-half).
#pragma unroll
    for (int mask = 1; mask < 16; mask <<= 1) {
#pragma unroll
        for (int v = 0; v < 8; ++v)
            partial[v] += __shfl_xor(partial[v], mask, 32);
    }
    if (l15 == 0) {
        const float bias2 = b2[0];
#pragma unroll
        for (int v = 0; v < 8; ++v) {
            int row = rowBase + half * 8 + v;
            if (row < n) scores[row] = partial[v] + bias2;
        }
    }
}

// ---------------------------------------------------------------------------
// Ordered-uint encoding of float for atomicMax (handles negatives).
// ---------------------------------------------------------------------------
__device__ __forceinline__ unsigned enc_f32(float f) {
    unsigned b = __float_as_uint(f);
    return (b >> 31) ? ~b : (b | 0x80000000u);
}
__device__ __forceinline__ float dec_f32(unsigned u) {
    unsigned b = (u >> 31) ? (u & 0x7fffffffu) : ~u;
    return __uint_as_float(b);
}

__global__ void k_init(float* __restrict__ out,
                       float* __restrict__ denom,
                       unsigned* __restrict__ segmax)
{
    int i = blockIdx.x * 256 + threadIdx.x;
    if (i < GG * DD) out[i] = 0.0f;
    if (i < GG) { denom[i] = 0.0f; segmax[i] = 0u; }
}

__global__ void k_segmax(const float* __restrict__ scores,
                         const long long* __restrict__ batch,
                         unsigned* __restrict__ segmax, int n)
{
    int i = blockIdx.x * 256 + threadIdx.x;
    if (i < n) {
        atomicMax(&segmax[(int)batch[i]], enc_f32(scores[i]));
    }
}

__global__ void k_exp(const float* __restrict__ scores,
                      const long long* __restrict__ batch,
                      const unsigned* __restrict__ segmax,
                      float* __restrict__ e,
                      float* __restrict__ denom, int n)
{
    int i = blockIdx.x * 256 + threadIdx.x;
    if (i < n) {
        int g = (int)batch[i];
        float m = dec_f32(segmax[g]);
        float v = __expf(scores[i] - m);
        e[i] = v;
        atomicAdd(&denom[g], v);
    }
}

// ---------------------------------------------------------------------------
// Pooling: out[g,:] += x[i,:] * e[i]/denom[g].  batch is sorted, so each
// block accumulates runs of equal g in registers and flushes one atomicAdd
// per (graph boundary, column). Thread t owns column t -> coalesced x reads.
// ---------------------------------------------------------------------------
__launch_bounds__(256)
__global__ void k_pool(const float* __restrict__ x,
                       const long long* __restrict__ batch,
                       const float* __restrict__ e,
                       const float* __restrict__ denom,
                       float* __restrict__ out, int n)
{
    const int t  = threadIdx.x;
    const int n0 = blockIdx.x * NPB;
    if (n0 >= n) return;
    const int n1 = (n0 + NPB < n) ? (n0 + NPB) : n;

    int gcur = (int)batch[n0];
    float inv = 1.0f / denom[gcur];
    float acc = 0.0f;

    for (int node = n0; node < n1; ++node) {
        int g = (int)batch[node];      // uniform across block
        if (g != gcur) {
            atomicAdd(&out[(size_t)gcur * DD + t], acc);
            acc  = 0.0f;
            gcur = g;
            inv  = 1.0f / denom[g];
        }
        if (node + 8 < n)
            __builtin_prefetch(&x[(size_t)(node + 8) * DD + t], 0, 0);
        float w = e[node] * inv;
        acc += x[(size_t)node * DD + t] * w;
    }
    atomicAdd(&out[(size_t)gcur * DD + t], acc);
}

// ---------------------------------------------------------------------------
// Inputs (reference order): x[N*D] f32, batch[N] i64, num_graphs (scalar),
// W1[D*D] f32, b1[D] f32, W2[D] f32, b2[1] f32.  Output: [G*D] f32.
// Workspace: scores[N] | e[N] | denom[G] | segmax[G] | Wt[8*256*40 bf16]
// ---------------------------------------------------------------------------
extern "C" void kernel_launch(void* const* d_in, const int* in_sizes, int n_in,
                              void* d_out, int out_size, void* d_ws, size_t ws_size,
                              hipStream_t stream)
{
    const float*     x     = (const float*)d_in[0];
    const long long* batch = (const long long*)d_in[1];
    const float*     W1    = (const float*)d_in[3];
    const float*     b1    = (const float*)d_in[4];
    const float*     W2    = (const float*)d_in[5];
    const float*     b2    = (const float*)d_in[6];
    float*           out   = (float*)d_out;

    const int n = in_sizes[1];   // N (node count)

    float*    scores = (float*)d_ws;
    float*    e      = scores + n;
    float*    denom  = e + n;
    unsigned* segmax = (unsigned*)(denom + GG);
    bf16*     Wt     = (bf16*)(segmax + GG);       // 16B-aligned (offset mult of 16)

    k_prep  <<<(256 * 256 + 255) / 256, 256, 0, stream>>>(W1, Wt);
    k_init  <<<(GG * DD + 255) / 256,   256, 0, stream>>>(out, denom, segmax);
    k_scores<<<(n + 127) / 128,         256, 0, stream>>>(x, Wt, b1, W2, b2, scores, n);
    k_segmax<<<(n + 255) / 256,         256, 0, stream>>>(scores, batch, segmax, n);
    k_exp   <<<(n + 255) / 256,         256, 0, stream>>>(scores, batch, segmax, e, denom, n);
    k_pool  <<<(n + NPB - 1) / NPB,     256, 0, stream>>>(x, batch, e, denom, out, n);
}
